// NeuralNetGlobalHammer_29712583753775
// MI455X (gfx1250) — matrix operand order repaired and checked by reading, not compiled, not run on hardware
//
#include <hip/hip_runtime.h>

typedef __attribute__((ext_vector_type(16))) _Float16 v16h;
typedef __attribute__((ext_vector_type(4)))  _Float16 v4h;
typedef __attribute__((ext_vector_type(8)))  float    v8f;
typedef __attribute__((ext_vector_type(4)))  float    v4f;

#define W_IN    16384
#define OW      16353          // W_IN - 31
#define NROWS   1024           // B*H = 16*64
#define CT      8192           // output positions per workgroup
#define NLOAD   (CT + 48)      // signal elements staged in LDS (halo + pad), mult of 4
#define SCALE_F 0.17782794100389228f   // sqrt(10^(-15/10))

__device__ __forceinline__ float fast_tanh(float x) {
  // tanh(x) = sign(x) * (1 - e)/(1 + e),  e = exp2(-2*log2(e)*|x|)  (no overflow)
  float ax = __builtin_fabsf(x);
  float e  = __builtin_amdgcn_exp2f(ax * -2.8853900817779268f);
  float r  = (1.0f - e) * __builtin_amdgcn_rcpf(1.0f + e);
  return __builtin_copysignf(r, x);
}

__global__ __launch_bounds__(256)
void NeuralNetGlobalHammer_kernel(const float* __restrict__ xr_g,
                                  const float* __restrict__ xi_g,
                                  const float* __restrict__ w1_g,   // (1,1,1,8)
                                  const float* __restrict__ w2_g,   // (1,1,8,1)
                                  const float* __restrict__ wr_g,   // (1,32,1,1)
                                  const float* __restrict__ wi_g,   // (1,32,1,1)
                                  float* __restrict__ out)          // [NROWS, OW, 2]
{
  __shared__ __align__(32) _Float16 smr[NLOAD];
  __shared__ __align__(32) _Float16 smi[NLOAD];

  const int row = blockIdx.y;
  const int c0  = blockIdx.x * CT;          // first output position of this chunk
  const int tid = threadIdx.x;

  // ---- MLP weights (broadcast loads, L2 resident) ----
  float w1[8], w2[8];
#pragma unroll
  for (int o = 0; o < 8; ++o) { w1[o] = w1_g[o]; w2[o] = w2_g[o]; }

  const float* xr_row = xr_g + (size_t)row * W_IN;
  const float* xi_row = xi_g + (size_t)row * W_IN;

  // ---- Phase 1: pointwise envelope MLP, re-attach phase, stage f16 to LDS ----
  // float4 loads / 8-byte packed f16 LDS stores (c0 and W_IN are multiples of 4,
  // so each vec4 is entirely in or entirely out of bounds).
  for (int i4 = tid * 4; i4 < NLOAD; i4 += 256 * 4) {
    const int g = c0 + i4;
    v4f xr4 = {0.f, 0.f, 0.f, 0.f}, xi4 = {0.f, 0.f, 0.f, 0.f};
    if (g < W_IN) {
      xr4 = *(const v4f*)(xr_row + g);
      xi4 = *(const v4f*)(xi_row + g);
    }
    v4h pr, pi;
#pragma unroll
    for (int j = 0; j < 4; ++j) {
      const float xr = xr4[j], xi = xi4[j];
      const float m2  = __builtin_fmaf(xr, xr, xi * xi);
      const float inv = __builtin_amdgcn_rsqf(__builtin_fmaxf(m2, 1e-30f)); // 1/|x|
      const float mag = m2 * inv;                                           // |x|
      float h = 0.0f;
#pragma unroll
      for (int o = 0; o < 8; ++o)
        h = __builtin_fmaf(w2[o], fast_tanh(w1[o] * mag), h);
      const float s = h * inv;     // h*cos = h*xr/|x| ; h*sin = h*xi/|x|
      pr[j] = (_Float16)(xr * s);
      pi[j] = (_Float16)(xi * s);
    }
    *(v4h*)(smr + i4) = pr;   // ds_store_b64
    *(v4h*)(smi + i4) = pi;   // ds_store_b64
  }
  __syncthreads();

  // ---- Phase 2: FIR as banded GEMM on WMMA ----
  // A (16x32 f16) layout per lane: M = lane&15, hi = lane>>4,
//   elem e -> K = e + 16*hi
  // A1[m,k] = w[k-m]    (k>=m) : taps 0..31-m
  // A2[m,k] = w[k+32-m] (k<m)  : taps 32-m..31, consumed against B shifted by +32
  const int lane = tid & 31;
  const int hi   = lane >> 4;
  const int MN   = lane & 15;     // row M for A, col N for B/D

  v16h A1r, A2r, A1i, A2i;
#pragma unroll
  for (int e = 0; e < 16; ++e) {
    const int K = e + 16 * hi;
    _Float16 a1r = (_Float16)0.f, a2r = (_Float16)0.f;
    _Float16 a1i = (_Float16)0.f, a2i = (_Float16)0.f;
    if (K >= MN) {
      a1r = (_Float16)wr_g[K - MN];
      a1i = (_Float16)wi_g[K - MN];
    } else {
      a2r = (_Float16)wr_g[K + 32 - MN];
      a2i = (_Float16)wi_g[K + 32 - MN];
    }
    A1r[e] = a1r; A2r[e] = a2r; A1i[e] = a1i; A2i[e] = a2i;
  }

  const int wave = tid >> 5;                    // 8 waves (wave32)
  float* out_row = out + (size_t)row * OW * 2;

#pragma unroll
  for (int it = 0; it < 4; ++it) {
    const int tt = wave * 4 + it;               // 32 tiles of 256 outputs per WG
    const int l0 = tt * 256;

    // B (32x16 f16) per lane: B[k,n] = s[t + 16n + k]; lane holds 16 contiguous
    // halfs starting at element l0 + 16*(N+hi)  -> 32B-aligned -> 2x ds_load_b128
    const int boff = l0 + 16 * (MN + hi);
    const v16h Br  = *(const v16h*)(smr + boff);
    const v16h Bpr = *(const v16h*)(smr + boff + 32);   // B at t+32 for A2
    const v16h Bi  = *(const v16h*)(smi + boff);
    const v16h Bpi = *(const v16h*)(smi + boff + 32);

    v8f frr = {}, fri = {}, fir_ = {}, fii = {};
    // fir(sr, wr)
    frr  = __builtin_amdgcn_wmma_f32_16x16x32_f16(false, A1r, false, Br,  (short)0, frr,  false, false);
    frr  = __builtin_amdgcn_wmma_f32_16x16x32_f16(false, A2r, false, Bpr, (short)0, frr,  false, false);
    // fir(sr, wi)
    fri  = __builtin_amdgcn_wmma_f32_16x16x32_f16(false, A1i, false, Br,  (short)0, fri,  false, false);
    fri  = __builtin_amdgcn_wmma_f32_16x16x32_f16(false, A2i, false, Bpr, (short)0, fri,  false, false);
    // fir(si, wr)
    fir_ = __builtin_amdgcn_wmma_f32_16x16x32_f16(false, A1r, false, Bi,  (short)0, fir_, false, false);
    fir_ = __builtin_amdgcn_wmma_f32_16x16x32_f16(false, A2r, false, Bpi, (short)0, fir_, false, false);
    // fir(si, wi)
    fii  = __builtin_amdgcn_wmma_f32_16x16x32_f16(false, A1i, false, Bi,  (short)0, fii,  false, false);
    fii  = __builtin_amdgcn_wmma_f32_16x16x32_f16(false, A2i, false, Bpi, (short)0, fii,  false, false);

    // D[m,n] = y[t + 16n + m]; this lane: n = MN, m = v + 8*hi
    const int p_base = c0 + l0 + 16 * MN + 8 * hi;
    // Wave-uniform tile bound, forced into an SGPR so the branch is scalar
    // (s_cmp/s_cbranch instead of v_cmpx + exec churn).
    const int tile_end = __builtin_amdgcn_readfirstlane(c0 + l0 + 256);
    if (tile_end <= OW) {
      // Fast path: whole tile in bounds -> unconditional stores.
#pragma unroll
      for (int v = 0; v < 8; ++v) {
        float2 val;
        val.x = SCALE_F * (frr[v] - fii[v]);   // yr
        val.y = SCALE_F * (fir_[v] + fri[v]);  // yi
        *(float2*)(out_row + (size_t)(p_base + v) * 2) = val;
      }
    } else {
      // Tail tile: per-element guard (only last tile of last chunk).
#pragma unroll
      for (int v = 0; v < 8; ++v) {
        const int p = p_base + v;
        if (p < OW) {
          float2 val;
          val.x = SCALE_F * (frr[v] - fii[v]);
          val.y = SCALE_F * (fir_[v] + fri[v]);
          *(float2*)(out_row + (size_t)p * 2) = val;
        }
      }
    }
  }
}

extern "C" void kernel_launch(void* const* d_in, const int* in_sizes, int n_in,
                              void* d_out, int out_size, void* d_ws, size_t ws_size,
                              hipStream_t stream) {
  (void)in_sizes; (void)n_in; (void)d_ws; (void)ws_size; (void)out_size;
  const float* xr = (const float*)d_in[0];
  const float* xi = (const float*)d_in[1];
  const float* w1 = (const float*)d_in[2];
  const float* w2 = (const float*)d_in[3];
  const float* wr = (const float*)d_in[4];
  const float* wi = (const float*)d_in[5];
  float* out = (float*)d_out;

  dim3 grid((OW + CT - 1) / CT, NROWS);  // (2, 1024)
  NeuralNetGlobalHammer_kernel<<<grid, 256, 0, stream>>>(xr, xi, w1, w2, wr, wi, out);
}
